// Attention_61701500174670
// MI455X (gfx1250) — matrix operand order repaired and checked by reading, not compiled, not run on hardware
//
#include <hip/hip_runtime.h>
#include <math.h>

// Problem constants (from reference): B=4, N=4096, D=512, fp32.
#define BATCH 4
#define SEQ   4096
#define HDIM  512
#define KT    256   // keys per stats iteration (2x to amortize softmax)
#define BM    32    // query rows per workgroup (2 row tiles of 16)
#define NW    8     // waves per workgroup (wave32, 256 threads)

typedef __attribute__((ext_vector_type(16))) __bf16 v16bf;
typedef __attribute__((ext_vector_type(8)))  float  v8f;

union V16 {
    v16bf v;
    float4 f4[2];
};

struct alignas(8)  bfx4 { __bf16 v[4]; };
struct alignas(16) bfx8 { __bf16 v[8]; };

__device__ __forceinline__ v8f wmma_bf16(v16bf a, v16bf b, v8f c) {
    // D = A(16x32 bf16) * B(32x16 bf16) + C(16x16 f32)
    return __builtin_amdgcn_wmma_f32_16x16x32_bf16(
        /*neg_a=*/false, a, /*neg_b=*/false, b,
        /*c_mod=*/(short)0, c, /*reuse_a=*/false, /*reuse_b=*/false);
}

// ---------------------------------------------------------------------------
// fp32 -> bf16 straight convert (for Q: [B][N][D] stays row-major in d)
// ---------------------------------------------------------------------------
__global__ __launch_bounds__(256)
void cvt_kernel(const float* __restrict__ in, __bf16* __restrict__ out, int n) {
    int i = (blockIdx.x * 256 + threadIdx.x) * 4;
    if (i >= n) return;
    float4 f = *(const float4*)(in + i);
    bfx4 o;
    o.v[0] = (__bf16)f.x; o.v[1] = (__bf16)f.y;
    o.v[2] = (__bf16)f.z; o.v[3] = (__bf16)f.w;
    *(bfx4*)(out + i) = o;
}

// ---------------------------------------------------------------------------
// fp32 [b][R][C] -> bf16 [b][C][R] tiled transpose-convert.
//   K: R=D, C=N  -> KbT [B][N][D]  (contraction d contiguous)
//   V: R=N, C=D  -> VbT [B][D][N]  (contraction n contiguous)
// ---------------------------------------------------------------------------
__global__ __launch_bounds__(128)
void tcvt_kernel(const float* __restrict__ in, __bf16* __restrict__ out,
                 int R, int C) {
    __shared__ __bf16 tile[32][40];   // +8 pad vs bank conflicts
    const int b  = blockIdx.z;
    const int c0 = blockIdx.x * 32;
    const int r0 = blockIdx.y * 32;
    const int t  = threadIdx.x;
    const int row = t >> 2;          // 0..31
    const int cg  = t & 3;           // 0..3 (8 cols each)

    const float* src = in + ((size_t)b * R + r0 + row) * C + c0 + cg * 8;
    float4 f0 = *(const float4*)(src);
    float4 f1 = *(const float4*)(src + 4);
    tile[row][cg*8+0] = (__bf16)f0.x; tile[row][cg*8+1] = (__bf16)f0.y;
    tile[row][cg*8+2] = (__bf16)f0.z; tile[row][cg*8+3] = (__bf16)f0.w;
    tile[row][cg*8+4] = (__bf16)f1.x; tile[row][cg*8+5] = (__bf16)f1.y;
    tile[row][cg*8+6] = (__bf16)f1.z; tile[row][cg*8+7] = (__bf16)f1.w;
    __syncthreads();

    const int ccol = t >> 2;         // output row (= input col)
    const int dg   = t & 3;          // group of 8 input rows
    bfx8 o;
    #pragma unroll
    for (int i = 0; i < 8; ++i) o.v[i] = tile[dg*8 + i][ccol];
    *(bfx8*)(out + ((size_t)b * C + c0 + ccol) * R + r0 + dg * 8) = o;
}

// ---------------------------------------------------------------------------
// Flash attention main kernel.
//  Block = 256 threads (8 waves), BM=32 query rows; every wave processes both
//  16-row tiles so each K/V B-operand feeds 2 WMMAs.
//  Scores: wave w owns key cols [w*32, w*32+32) of the 256-key tile.
//  PV:     wave w owns output cols [w*64, w*64+64).
//  Global B-operands are explicitly software-pipelined (double buffered).
// ---------------------------------------------------------------------------
__global__ __launch_bounds__(256)
void fa_kernel(const __bf16* __restrict__ Qb,   // [B][N][D]
               const __bf16* __restrict__ KbT,  // [B][N][D] (K transposed)
               const __bf16* __restrict__ VbT,  // [B][D][N] (V transposed)
               float* __restrict__ out) {       // [B][N][D] fp32
    __shared__ __bf16 q_lds[BM * HDIM];   // 32 KB
    __shared__ __bf16 p_lds[BM * KT];     // 16 KB
    __shared__ float  redmax[NW * BM];
    __shared__ float  redsum[NW * BM];

    const int tid = threadIdx.x;
    const int w   = tid >> 5;       // wave 0..7
    const int l   = tid & 31;       // lane
    const int ln  = l & 15;         // lane within half
    const int h   = l >> 4;         // half: 0 or 1
    const int bb  = blockIdx.y;     // batch
    const int row0 = blockIdx.x * BM;

    // ---- stage Q tile (32 x 512 bf16) into LDS, row-major in d ----
    {
        const __bf16* src = Qb + ((size_t)bb * SEQ + row0) * HDIM;
        for (int i = tid; i < BM * HDIM / 8; i += 256)
            *(float4*)(q_lds + i * 8) = *(const float4*)(src + i * 8);
    }
    __syncthreads();

    // ---- accumulators: O[row tile r][col tile ct] ----
    v8f O[2][4];
    #pragma unroll
    for (int r = 0; r < 2; ++r)
        #pragma unroll
        for (int ct = 0; ct < 4; ++ct)
            #pragma unroll
            for (int e = 0; e < 8; ++e) O[r][ct][e] = 0.f;
    float m_run[2][8], l_run[2][8];
    #pragma unroll
    for (int r = 0; r < 2; ++r)
        #pragma unroll
        for (int j = 0; j < 8; ++j) { m_run[r][j] = -INFINITY; l_run[r][j] = 0.f; }

    // Hoisted base pointers for this wave/lane
    const __bf16* kbase = KbT + ((size_t)bb * SEQ + w * 32 + ln) * HDIM + 16 * h;
    const __bf16* vbase = VbT + ((size_t)bb * HDIM + w * 64 + ln) * SEQ + 16 * h;

    for (int kt = 0; kt < SEQ / KT; ++kt) {
        // ====== scores: S[r][s](16x16) = Q[r] @ K slice (s = key subtile) ===
        v8f S[2][2];
        #pragma unroll
        for (int r = 0; r < 2; ++r)
            #pragma unroll
            for (int s = 0; s < 2; ++s)
                #pragma unroll
                for (int e = 0; e < 8; ++e) S[r][s][e] = 0.f;

        const __bf16* kp0 = kbase + (size_t)kt * KT * HDIM;        // key subtile 0
        const __bf16* kp1 = kp0 + (size_t)16 * HDIM;               // key subtile 1

        V16 bk0, bk1;                           // current K B-operands
        bk0.f4[0] = *(const float4*)(kp0);
        bk0.f4[1] = *(const float4*)(kp0 + 8);
        bk1.f4[0] = *(const float4*)(kp1);
        bk1.f4[1] = *(const float4*)(kp1 + 8);

        #pragma unroll
        for (int d0 = 0; d0 < HDIM; d0 += 32) {
            V16 nk0, nk1;                       // prefetch next d-step
            if (d0 + 32 < HDIM) {
                nk0.f4[0] = *(const float4*)(kp0 + d0 + 32);
                nk0.f4[1] = *(const float4*)(kp0 + d0 + 40);
                nk1.f4[0] = *(const float4*)(kp1 + d0 + 32);
                nk1.f4[1] = *(const float4*)(kp1 + d0 + 40);
            }
            V16 a[2];  // A layout: lane ln = row, k runs [16h,+16)
            #pragma unroll
            for (int r = 0; r < 2; ++r) {
                a[r].f4[0] = *(const float4*)(q_lds + (r * 16 + ln) * HDIM + d0 + 16 * h);
                a[r].f4[1] = *(const float4*)(q_lds + (r * 16 + ln) * HDIM + d0 + 16 * h + 8);
            }
            #pragma unroll
            for (int r = 0; r < 2; ++r) {
                S[r][0] = wmma_bf16(a[r].v, bk0.v, S[r][0]);
                S[r][1] = wmma_bf16(a[r].v, bk1.v, S[r][1]);
            }
            if (d0 + 32 < HDIM) { bk0 = nk0; bk1 = nk1; }
        }

        // Early-issue first PV V-chunk; completes under the softmax work.
        const __bf16* vp = vbase + kt * KT;
        V16 BV[4];
        #pragma unroll
        for (int ct = 0; ct < 4; ++ct) {
            BV[ct].f4[0] = *(const float4*)(vp + (size_t)ct * 16 * SEQ);
            BV[ct].f4[1] = *(const float4*)(vp + (size_t)ct * 16 * SEQ + 8);
        }

        // ====== online softmax =============================================
        // C layout: VGPR j <-> row (r*16 + j + 8h), lane ln <-> key col.
        float rmax[2][8];
        #pragma unroll
        for (int r = 0; r < 2; ++r)
            #pragma unroll
            for (int j = 0; j < 8; ++j)
                rmax[r][j] = fmaxf(S[r][0][j], S[r][1][j]);
        #pragma unroll
        for (int m = 1; m < 16; m <<= 1)
            #pragma unroll
            for (int r = 0; r < 2; ++r)
                #pragma unroll
                for (int j = 0; j < 8; ++j)
                    rmax[r][j] = fmaxf(rmax[r][j], __shfl_xor(rmax[r][j], m, 32));
        if (ln == 0) {
            #pragma unroll
            for (int r = 0; r < 2; ++r)
                #pragma unroll
                for (int j = 0; j < 8; ++j)
                    redmax[w * BM + r * 16 + 8 * h + j] = rmax[r][j];
        }
        __syncthreads();

        #pragma unroll
        for (int r = 0; r < 2; ++r) {
            float scale[8];
            #pragma unroll
            for (int j = 0; j < 8; ++j) {
                float mt = redmax[0 * BM + r * 16 + 8 * h + j];
                #pragma unroll
                for (int ww = 1; ww < NW; ++ww)
                    mt = fmaxf(mt, redmax[ww * BM + r * 16 + 8 * h + j]);
                float mn = fmaxf(m_run[r][j], mt);
                scale[j] = __expf(m_run[r][j] - mn);
                m_run[r][j] = mn;
                l_run[r][j] *= scale[j];
            }
            #pragma unroll
            for (int ct = 0; ct < 4; ++ct)
                #pragma unroll
                for (int j = 0; j < 8; ++j) O[r][ct][j] *= scale[j];
        }

        // P = exp(S - m): write bf16 tile to LDS, accumulate row sums
        float rsum[2][8];
        #pragma unroll
        for (int r = 0; r < 2; ++r)
            #pragma unroll
            for (int j = 0; j < 8; ++j) {
                float p0 = __expf(S[r][0][j] - m_run[r][j]);
                float p1 = __expf(S[r][1][j] - m_run[r][j]);
                rsum[r][j] = p0 + p1;
                p_lds[(r * 16 + 8 * h + j) * KT + w * 32 + ln]      = (__bf16)p0;
                p_lds[(r * 16 + 8 * h + j) * KT + w * 32 + 16 + ln] = (__bf16)p1;
            }
        #pragma unroll
        for (int m = 1; m < 16; m <<= 1)
            #pragma unroll
            for (int r = 0; r < 2; ++r)
                #pragma unroll
                for (int j = 0; j < 8; ++j)
                    rsum[r][j] += __shfl_xor(rsum[r][j], m, 32);
        if (ln == 0) {
            #pragma unroll
            for (int r = 0; r < 2; ++r)
                #pragma unroll
                for (int j = 0; j < 8; ++j)
                    redsum[w * BM + r * 16 + 8 * h + j] = rsum[r][j];
        }
        __syncthreads();
        #pragma unroll
        for (int r = 0; r < 2; ++r)
            #pragma unroll
            for (int j = 0; j < 8; ++j) {
                float s = 0.f;
                #pragma unroll
                for (int ww = 0; ww < NW; ++ww)
                    s += redsum[ww * BM + r * 16 + 8 * h + j];
                l_run[r][j] += s;
            }

        // ====== O[r] += P[r](16x256) @ V(256x64 slice), double buffered ====
        #pragma unroll
        for (int q = 0; q < KT / 32; ++q) {     // 8 chunks of 32 keys
            V16 BN[4];                          // prefetch next chunk's V
            if (q + 1 < KT / 32) {
                #pragma unroll
                for (int ct = 0; ct < 4; ++ct) {
                    BN[ct].f4[0] = *(const float4*)(vp + (size_t)ct * 16 * SEQ + 32 * (q + 1));
                    BN[ct].f4[1] = *(const float4*)(vp + (size_t)ct * 16 * SEQ + 32 * (q + 1) + 8);
                }
            }
            V16 A[2];
            #pragma unroll
            for (int r = 0; r < 2; ++r) {
                A[r].f4[0] = *(const float4*)(p_lds + (r * 16 + ln) * KT + 32 * q + 16 * h);
                A[r].f4[1] = *(const float4*)(p_lds + (r * 16 + ln) * KT + 32 * q + 16 * h + 8);
            }
            #pragma unroll
            for (int ct = 0; ct < 4; ++ct)
                #pragma unroll
                for (int r = 0; r < 2; ++r)
                    O[r][ct] = wmma_bf16(A[r].v, BV[ct].v, O[r][ct]);
            if (q + 1 < KT / 32) {
                #pragma unroll
                for (int ct = 0; ct < 4; ++ct) BV[ct] = BN[ct];
            }
        }
        __syncthreads();   // protect p_lds / red arrays before next tile
    }

    // ---- epilogue: normalize by row sums, store fp32 ----
    #pragma unroll
    for (int r = 0; r < 2; ++r) {
        float inv[8];
        #pragma unroll
        for (int j = 0; j < 8; ++j) inv[j] = 1.f / l_run[r][j];
        #pragma unroll
        for (int ct = 0; ct < 4; ++ct) {
            #pragma unroll
            for (int j = 0; j < 8; ++j) {
                const int row = row0 + r * 16 + 8 * h + j;
                const int d   = w * 64 + ct * 16 + ln;
                out[((size_t)bb * SEQ + row) * HDIM + d] = O[r][ct][j] * inv[j];
            }
        }
    }
}

// ---------------------------------------------------------------------------
extern "C" void kernel_launch(void* const* d_in, const int* in_sizes, int n_in,
                              void* d_out, int out_size, void* d_ws, size_t ws_size,
                              hipStream_t stream) {
    const float* Q = (const float*)d_in[0];   // [B][N][D]
    const float* K = (const float*)d_in[1];   // [B][D][N]
    const float* V = (const float*)d_in[2];   // [B][N][D]
    float* out = (float*)d_out;

    const size_t elems = (size_t)BATCH * SEQ * HDIM;  // 8.4M per tensor
    __bf16* Qb  = (__bf16*)d_ws;
    __bf16* KbT = Qb  + elems;
    __bf16* VbT = KbT + elems;

    // 1) convert Q to bf16 (layout unchanged)
    cvt_kernel<<<(int)(elems / 4 / 256), 256, 0, stream>>>(Q, Qb, (int)elems);
    // 2) K [b][D][N] fp32 -> KbT [b][N][D] bf16 (d contiguous per key)
    dim3 gk(SEQ / 32, HDIM / 32, BATCH);
    tcvt_kernel<<<gk, 128, 0, stream>>>(K, KbT, HDIM, SEQ);
    // 3) V [b][N][D] fp32 -> VbT [b][D][N] bf16 (n contiguous per d)
    dim3 gv(HDIM / 32, SEQ / 32, BATCH);
    tcvt_kernel<<<gv, 128, 0, stream>>>(V, VbT, SEQ, HDIM);
    // 4) flash attention
    dim3 g(SEQ / BM, BATCH);
    fa_kernel<<<g, 256, 0, stream>>>(Qb, KbT, VbT, out);
}
